// PerceiverAttention_46299747451555
// MI455X (gfx1250) — matrix operand order repaired and checked by reading, not compiled, not run on hardware
//
#include <hip/hip_runtime.h>
#include <hip/hip_bf16.h>
#include <math.h>

#define DIMc   1024
#define Hc     16
#define Dhc    64
#define INNERc 1024
#define Bc     4
#define Nc     256
#define Sc     8192

typedef __attribute__((ext_vector_type(16))) __bf16 v16bf;
typedef __attribute__((ext_vector_type(8)))  __bf16 v8bf;
typedef __attribute__((ext_vector_type(4)))  __bf16 v4bf;
typedef __attribute__((ext_vector_type(8)))  float  v8f;

union Frag16 { v16bf v; uint4 u[2]; };

// A-matrix 16x32 bf16 fragment: lane m = lane&15, K chunks {b..b+7, b+16..b+23}, b=(lane>>4)*8
__device__ __forceinline__ v16bf load_a_frag(const __bf16* rowbase, int k0, int lane) {
  int base = (lane >> 4) * 8;
  Frag16 f;
  f.u[0] = *reinterpret_cast<const uint4*>(rowbase + k0 + base);
  f.u[1] = *reinterpret_cast<const uint4*>(rowbase + k0 + base + 16);
  return f.v;
}
// B-matrix 32x16 bf16 fragment (rowbase = column n's K-run): K chunks {b..b+7, b+16..b+23}, b=(lane>>4)*8
__device__ __forceinline__ v16bf load_b_frag(const __bf16* rowbase, int k0, int lane) {
  int start = k0 + (lane >> 4) * 8;
  Frag16 f;
  f.u[0] = *reinterpret_cast<const uint4*>(rowbase + start);
  f.u[1] = *reinterpret_cast<const uint4*>(rowbase + start + 16);
  return f.v;
}
__device__ __forceinline__ v8f wmma_bf16(v16bf a, v16bf b, v8f c) {
  return __builtin_amdgcn_wmma_f32_16x16x32_bf16(false, a, false, b, (short)0, c, false, false);
}

// Async global->LDS copy: each lane moves 16 bytes; tracked by ASYNCcnt.
__device__ __forceinline__ void async_copy_b128(void* lds_dst, const void* gsrc) {
  unsigned lds_off = (unsigned)(uintptr_t)lds_dst;   // LDS aperture: addr[31:0] == LDS byte offset
  asm volatile("global_load_async_to_lds_b128 %0, %1, off"
               :: "v"(lds_off), "v"(gsrc) : "memory");
}
__device__ __forceinline__ void wait_asynccnt0() {
  asm volatile("s_wait_asynccnt 0" ::: "memory");
}

// ---------------- LayerNorm (fp32 in) + cast to bf16 ----------------
__global__ __launch_bounds__(256) void ln_cast_kernel(const float* __restrict__ x,
                                                      const float* __restrict__ w,
                                                      const float* __restrict__ bias,
                                                      __bf16* __restrict__ y) {
  __shared__ float ssum[256], ssq[256];
  int row = blockIdx.x, tid = threadIdx.x;
  const float4 v = reinterpret_cast<const float4*>(x + (size_t)row * DIMc)[tid];
  ssum[tid] = v.x + v.y + v.z + v.w;
  ssq[tid]  = v.x * v.x + v.y * v.y + v.z * v.z + v.w * v.w;
  __syncthreads();
  for (int off = 128; off > 0; off >>= 1) {
    if (tid < off) { ssum[tid] += ssum[tid + off]; ssq[tid] += ssq[tid + off]; }
    __syncthreads();
  }
  float mean = ssum[0] * (1.0f / DIMc);
  float var  = ssq[0] * (1.0f / DIMc) - mean * mean;
  float rstd = rsqrtf(var + 1e-5f);
  const float4 wv = reinterpret_cast<const float4*>(w)[tid];
  const float4 bv = reinterpret_cast<const float4*>(bias)[tid];
  v4bf o;
  o[0] = (__bf16)((v.x - mean) * rstd * wv.x + bv.x);
  o[1] = (__bf16)((v.y - mean) * rstd * wv.y + bv.y);
  o[2] = (__bf16)((v.z - mean) * rstd * wv.z + bv.z);
  o[3] = (__bf16)((v.w - mean) * rstd * wv.w + bv.w);
  reinterpret_cast<v4bf*>(y + (size_t)row * DIMc)[tid] = o;
}

// ---------------- Transpose + cast weights: in [K][N] fp32 -> out [N][K] bf16 ----------------
__global__ __launch_bounds__(256) void transpose_cast_kernel(const float* __restrict__ in,
                                                             __bf16* __restrict__ out,
                                                             int K, int N) {
  int idx = blockIdx.x * 256 + threadIdx.x;
  int n = idx / K, k = idx - n * K;
  out[idx] = (__bf16)in[(size_t)k * N + n];
}

// ---------------- WMMA GEMM: C[M,N] = A[M,K] @ Bt[N,K]^T (+bias if F32OUT) ----------------
// Block tile 64x256, 8 waves as 2(M) x 4(N); each wave: 32x64 (8 accumulators).
// A tile (64x32 bf16 = 4KB) double-buffered in LDS via GLOBAL_LOAD_ASYNC_TO_LDS_B128.
template <bool F32OUT>
__global__ __launch_bounds__(256) void gemm_wmma_kernel(const __bf16* __restrict__ A,
                                                        const __bf16* __restrict__ Bt,
                                                        __bf16* __restrict__ Cb,
                                                        float* __restrict__ Cf,
                                                        const float* __restrict__ bias,
                                                        int M, int N, int K) {
  __shared__ __bf16 ldsA[2][64 * 32];
  int tid = threadIdx.x, lane = tid & 31, wave = tid >> 5;
  int ln = lane & 15, hl = lane >> 4;
  int wm = wave >> 2, wn = wave & 3;
  int row0g = blockIdx.y * 64;                 // block's first global row
  int col0  = blockIdx.x * 256 + wn * 64;      // wave's first column

  // staging assignment: thread t copies 16B chunk (row = t>>2, kchunk = t&3)
  int srow = tid >> 2, schunk = (tid & 3) * 8;
  const __bf16* sgp = A + (size_t)(row0g + srow) * K + schunk;

  v8f acc[2][4] = {};

  // prologue: stage k0 = 0 into buffer 0
  async_copy_b128(&ldsA[0][srow * 32 + schunk], sgp);

  const __bf16* brow[4];
#pragma unroll
  for (int nt = 0; nt < 4; ++nt) brow[nt] = Bt + (size_t)(col0 + nt * 16 + ln) * K;

  for (int k0 = 0; k0 < K; k0 += 32) {
    int cur = (k0 >> 5) & 1;
    wait_asynccnt0();
    __syncthreads();
    if (k0 + 32 < K)  // overlap next-tile copy with this tile's WMMAs
      async_copy_b128(&ldsA[cur ^ 1][srow * 32 + schunk], sgp + k0 + 32);

    v16bf af0 = load_a_frag(&ldsA[cur][(wm * 32 + ln) * 32], 0, lane);
    v16bf af1 = load_a_frag(&ldsA[cur][(wm * 32 + 16 + ln) * 32], 0, lane);
#pragma unroll
    for (int nt = 0; nt < 4; ++nt) {
      if (k0 + 32 < K) __builtin_prefetch(brow[nt] + k0 + 64, 0, 0);
      v16bf bf = load_b_frag(brow[nt], k0, lane);
      acc[0][nt] = wmma_bf16(af0, bf, acc[0][nt]);
      acc[1][nt] = wmma_bf16(af1, bf, acc[1][nt]);
    }
  }

#pragma unroll
  for (int sm = 0; sm < 2; ++sm)
#pragma unroll
    for (int nt = 0; nt < 4; ++nt) {
      int col = col0 + nt * 16 + ln;
#pragma unroll
      for (int r = 0; r < 8; ++r) {
        int rowm = row0g + wm * 32 + sm * 16 + r + 8 * hl;
        if constexpr (F32OUT) {
          Cf[(size_t)rowm * N + col] = acc[sm][nt][r] + bias[col];
        } else {
          Cb[(size_t)rowm * N + col] = (__bf16)acc[sm][nt][r];
        }
      }
    }
}

// ---------------- Flash attention: per wave 16 query rows, stream S in 32-wide tiles ----------------
__global__ __launch_bounds__(256) void attention_kernel(const __bf16* __restrict__ q,
                                                        const __bf16* __restrict__ kv,
                                                        __bf16* __restrict__ o_out) {
  __shared__ __bf16 ldsVt[64 * 32];        // V tile transposed: [d][s]
  __shared__ __bf16 ldsP[8][16 * 32];      // per-wave P tile: [m][s]
  int tid = threadIdx.x, lane = tid & 31, wave = tid >> 5;
  int ln = lane & 15, hl = lane >> 4;
  int id = blockIdx.x >> 1;
  int b = id >> 4, h = id & 15;
  int n0 = (blockIdx.x & 1) * 128 + wave * 16;

  const __bf16* qrow = q + (size_t)(b * Nc + n0 + ln) * INNERc + h * Dhc;
  v16bf Aq0 = load_a_frag(qrow, 0, lane);
  v16bf Aq1 = load_a_frag(qrow, 32, lane);

  v8f o0 = {}, o1 = {}, o2 = {}, o3 = {};
  float mrow[8], lrow[8];
#pragma unroll
  for (int r = 0; r < 8; ++r) { mrow[r] = -3.0e38f; lrow[r] = 0.0f; }

  const size_t kvb = (size_t)b * Sc * (2 * INNERc);

  for (int s0 = 0; s0 < Sc; s0 += 32) {
    __syncthreads();
    {  // cooperatively stage V tile transposed into LDS
      int si = tid >> 3, d0 = (tid & 7) * 8;
      const __bf16* vp = kv + kvb + (size_t)(s0 + si) * (2 * INNERc) + INNERc + h * Dhc + d0;
      v8bf vv = *reinterpret_cast<const v8bf*>(vp);
#pragma unroll
      for (int i = 0; i < 8; ++i) ldsVt[(d0 + i) * 32 + si] = vv[i];
    }
    __syncthreads();

    // scores: 16 rows x 32 s-cols, K read straight from kv (contiguous in d)
    v8f sc[2];
#pragma unroll
    for (int ns = 0; ns < 2; ++ns) {
      v8f a = {};
      const __bf16* krow = kv + kvb + (size_t)(s0 + ns * 16 + ln) * (2 * INNERc) + h * Dhc;
      a = wmma_bf16(Aq0, load_b_frag(krow, 0, lane), a);
      a = wmma_bf16(Aq1, load_b_frag(krow, 32, lane), a);
#pragma unroll
      for (int r = 0; r < 8; ++r) {
        float x = a[r] * 0.125f;
        a[r] = fminf(11.0f, fmaxf(-11.0f, x));
      }
      sc[ns] = a;
    }

    // row max over 32 cols (butterfly within each 16-lane half; row r lives in VGPR r)
    float t[8];
#pragma unroll
    for (int r = 0; r < 8; ++r) t[r] = fmaxf(sc[0][r], sc[1][r]);
#pragma unroll
    for (int m = 1; m <= 8; m <<= 1)
#pragma unroll
      for (int r = 0; r < 8; ++r) t[r] = fmaxf(t[r], __shfl_xor(t[r], m, 32));

    float alpha[8], rs[8];
#pragma unroll
    for (int r = 0; r < 8; ++r) {
      float mn = fmaxf(mrow[r], t[r]);
      alpha[r] = __expf(mrow[r] - mn);
      mrow[r] = mn;
    }
#pragma unroll
    for (int ns = 0; ns < 2; ++ns)
#pragma unroll
      for (int r = 0; r < 8; ++r) sc[ns][r] = __expf(sc[ns][r] - mrow[r]);
#pragma unroll
    for (int r = 0; r < 8; ++r) rs[r] = sc[0][r] + sc[1][r];
#pragma unroll
    for (int m = 1; m <= 8; m <<= 1)
#pragma unroll
      for (int r = 0; r < 8; ++r) rs[r] += __shfl_xor(rs[r], m, 32);
#pragma unroll
    for (int r = 0; r < 8; ++r) {
      lrow[r] = lrow[r] * alpha[r] + rs[r];
      o0[r] *= alpha[r]; o1[r] *= alpha[r]; o2[r] *= alpha[r]; o3[r] *= alpha[r];
    }

    // bounce P through LDS: C layout -> A layout
    __bf16* pw = &ldsP[wave][0];
#pragma unroll
    for (int ns = 0; ns < 2; ++ns)
#pragma unroll
      for (int r = 0; r < 8; ++r)
        pw[(r + 8 * hl) * 32 + ns * 16 + ln] = (__bf16)sc[ns][r];
    asm volatile("s_wait_dscnt 0" ::: "memory");

    v16bf pf = load_a_frag(pw + (size_t)ln * 32, 0, lane);
    o0 = wmma_bf16(pf, load_b_frag(&ldsVt[( 0 + ln) * 32], 0, lane), o0);
    o1 = wmma_bf16(pf, load_b_frag(&ldsVt[(16 + ln) * 32], 0, lane), o1);
    o2 = wmma_bf16(pf, load_b_frag(&ldsVt[(32 + ln) * 32], 0, lane), o2);
    o3 = wmma_bf16(pf, load_b_frag(&ldsVt[(48 + ln) * 32], 0, lane), o3);
  }

#pragma unroll
  for (int r = 0; r < 8; ++r) {
    float inv = 1.0f / lrow[r];
    size_t orow = (size_t)(b * Nc + n0 + r + 8 * hl) * INNERc + h * Dhc;
    o_out[orow +  0 + ln] = (__bf16)(o0[r] * inv);
    o_out[orow + 16 + ln] = (__bf16)(o1[r] * inv);
    o_out[orow + 32 + ln] = (__bf16)(o2[r] * inv);
    o_out[orow + 48 + ln] = (__bf16)(o3[r] * inv);
  }
}

// ---------------- host launcher ----------------
extern "C" void kernel_launch(void* const* d_in, const int* in_sizes, int n_in,
                              void* d_out, int out_size, void* d_ws, size_t ws_size,
                              hipStream_t stream) {
  (void)in_sizes; (void)n_in; (void)out_size; (void)ws_size;
  const float* latents = (const float*)d_in[0];
  const float* context = (const float*)d_in[1];
  const float* ln_l_w  = (const float*)d_in[2];
  const float* ln_l_b  = (const float*)d_in[3];
  const float* ln_c_w  = (const float*)d_in[4];
  const float* ln_c_b  = (const float*)d_in[5];
  const float* Wq      = (const float*)d_in[6];
  const float* Wkv     = (const float*)d_in[7];
  const float* Wo      = (const float*)d_in[8];
  const float* bo      = (const float*)d_in[9];
  float* out = (float*)d_out;

  char* ws = (char*)d_ws;
  size_t off = 0;
  auto alloc = [&](size_t bytes) {
    void* p = ws + off;
    off = (off + bytes + 255) & ~(size_t)255;
    return p;
  };
  __bf16* lat_ln  = (__bf16*)alloc((size_t)Bc * Nc * DIMc * 2);
  __bf16* ctx_ln  = (__bf16*)alloc((size_t)Bc * Sc * DIMc * 2);
  __bf16* WqT     = (__bf16*)alloc((size_t)INNERc * DIMc * 2);
  __bf16* WkvT    = (__bf16*)alloc((size_t)2 * INNERc * DIMc * 2);
  __bf16* WoT     = (__bf16*)alloc((size_t)DIMc * INNERc * 2);
  __bf16* qb      = (__bf16*)alloc((size_t)Bc * Nc * INNERc * 2);
  __bf16* kvb     = (__bf16*)alloc((size_t)Bc * Sc * 2 * INNERc * 2);
  __bf16* attnout = (__bf16*)alloc((size_t)Bc * Nc * INNERc * 2);

  // 1) layernorm + cast
  ln_cast_kernel<<<Bc * Nc, 256, 0, stream>>>(latents, ln_l_w, ln_l_b, lat_ln);
  ln_cast_kernel<<<Bc * Sc, 256, 0, stream>>>(context, ln_c_w, ln_c_b, ctx_ln);

  // 2) weight transpose + cast:  W [K][N] -> Wt [N][K] bf16
  transpose_cast_kernel<<<(INNERc * DIMc) / 256, 256, 0, stream>>>(Wq, WqT, DIMc, INNERc);
  transpose_cast_kernel<<<(2 * INNERc * DIMc) / 256, 256, 0, stream>>>(Wkv, WkvT, DIMc, 2 * INNERc);
  transpose_cast_kernel<<<(DIMc * INNERc) / 256, 256, 0, stream>>>(Wo, WoT, INNERc, DIMc);

  // 3) Q = lat_ln @ Wq ;  KV = ctx_ln @ Wkv
  dim3 gq(INNERc / 256, (Bc * Nc) / 64);
  gemm_wmma_kernel<false><<<gq, 256, 0, stream>>>(lat_ln, WqT, qb, nullptr, nullptr,
                                                  Bc * Nc, INNERc, DIMc);
  dim3 gkv((2 * INNERc) / 256, (Bc * Sc) / 64);
  gemm_wmma_kernel<false><<<gkv, 256, 0, stream>>>(ctx_ln, WkvT, kvb, nullptr, nullptr,
                                                   Bc * Sc, 2 * INNERc, DIMc);

  // 4) flash attention: 2 blocks per (b,h), 8 waves x 16 rows each
  attention_kernel<<<Bc * Hc * 2, 256, 0, stream>>>(qb, kvb, attnout);

  // 5) out = attnout @ Wo + bo  (fp32 output)
  dim3 go(DIMc / 256, (Bc * Nc) / 64);
  gemm_wmma_kernel<true><<<go, 256, 0, stream>>>(attnout, WoT, nullptr, out, bo,
                                                 Bc * Nc, DIMc, INNERc);
}
